// NNUE_7748121002128
// MI455X (gfx1250) — matrix operand order repaired and checked by reading, not compiled, not run on hardware
//
#include <hip/hip_runtime.h>

typedef __attribute__((ext_vector_type(16))) __bf16 v16bf;
typedef __attribute__((ext_vector_type(8)))  __bf16 v8bf;
typedef __attribute__((ext_vector_type(8)))  float  v8f;

static constexpr int D_IN  = 40960;
static constexpr int BATCH = 2048;
static constexpr int NBASE = 288;
static constexpr int NCAT  = 576;   // 2*NBASE
static constexpr int KPAD  = 736;   // 576 + 144 + 16 pad  (23*32)
static constexpr int NMOVE = 4096;
static constexpr int NA0P  = 160;   // 144 padded to 5*32

// ---------------------------------------------------------------------------
// Transpose W[288][40960] (row-major) -> Wt[40960][288] so that the sparse
// accumulator gathers contiguous 1152B rows instead of stride-160KB columns.
// ---------------------------------------------------------------------------
__global__ __launch_bounds__(256)
void transpose_kernel(const float* __restrict__ W, float* __restrict__ Wt) {
  __shared__ float tile[32][33];
  const int jb = blockIdx.x * 32;           // D_IN dimension
  const int ib = blockIdx.y * 32;           // NBASE dimension
  const int tx = threadIdx.x, ty = threadIdx.y;
  #pragma unroll
  for (int r = 0; r < 4; ++r) {
    int i = ib + ty + r * 8;
    tile[ty + r * 8][tx] = W[(size_t)i * D_IN + jb + tx];
  }
  __syncthreads();
  #pragma unroll
  for (int r = 0; r < 4; ++r) {
    int j = jb + ty + r * 8;
    Wt[(size_t)j * NBASE + ib + tx] = tile[tx][ty + r * 8];
  }
}

// ---------------------------------------------------------------------------
// Pack Wcat (bf16 [4096,736]) = [ W_as | W_a1 | 0 ] along K.
// ---------------------------------------------------------------------------
__global__ __launch_bounds__(256)
void pack_cat_kernel(const float* __restrict__ W_as, const float* __restrict__ W_a1,
                     __bf16* __restrict__ Wcat) {
  size_t i = (size_t)blockIdx.x * 256 + threadIdx.x;
  if (i >= (size_t)NMOVE * KPAD) return;
  int n = (int)(i / KPAD), c = (int)(i % KPAD);
  float v = 0.0f;
  if (c < NCAT)            v = W_as[(size_t)n * NCAT + c];
  else if (c < NCAT + 144) v = W_a1[(size_t)n * 144 + (c - NCAT)];
  Wcat[i] = (__bf16)v;
}

// Pack W_a0 (f32 [144,576]) -> bf16 [160,576], rows 144..159 zero.
__global__ __launch_bounds__(256)
void pack_wa0_kernel(const float* __restrict__ W_a0, __bf16* __restrict__ Wa0b) {
  size_t i = (size_t)blockIdx.x * 256 + threadIdx.x;
  if (i >= (size_t)NA0P * NCAT) return;
  int n = (int)(i / NCAT), c = (int)(i % NCAT);
  Wa0b[i] = (n < 144) ? (__bf16)W_a0[(size_t)n * NCAT + c] : (__bf16)0.0f;
}

// ---------------------------------------------------------------------------
// Sparse half-kp accumulator: one block per batch row.
// ---------------------------------------------------------------------------
__global__ __launch_bounds__(320)
void accum_kernel(const float* __restrict__ pov,
                  const float* __restrict__ white, const float* __restrict__ black,
                  const float* __restrict__ WtW, const float* __restrict__ b_w,
                  const float* __restrict__ WtB, const float* __restrict__ b_b,
                  float* __restrict__ baseF, __bf16* __restrict__ Xbf) {
  const int r = blockIdx.x;
  const int tid = threadIdx.x;
  __shared__ int   cnt;
  __shared__ int   idxs[1024];
  __shared__ float vals[1024];

  float accW = 0.0f, accB = 0.0f;

  #pragma unroll 1
  for (int pass = 0; pass < 2; ++pass) {
    const float* src = pass == 0 ? white : black;
    const float* Wt  = pass == 0 ? WtW : WtB;
    const float* bia = pass == 0 ? b_w : b_b;

    if (tid == 0) cnt = 0;
    __syncthreads();
    const float4* row = (const float4*)(src + (size_t)r * D_IN);
    for (int j4 = tid; j4 < D_IN / 4; j4 += 320) {
      float4 v = row[j4];
      if (v.x != 0.0f) { int p = atomicAdd(&cnt, 1); if (p < 1024) { idxs[p] = 4*j4+0; vals[p] = v.x; } }
      if (v.y != 0.0f) { int p = atomicAdd(&cnt, 1); if (p < 1024) { idxs[p] = 4*j4+1; vals[p] = v.y; } }
      if (v.z != 0.0f) { int p = atomicAdd(&cnt, 1); if (p < 1024) { idxs[p] = 4*j4+2; vals[p] = v.z; } }
      if (v.w != 0.0f) { int p = atomicAdd(&cnt, 1); if (p < 1024) { idxs[p] = 4*j4+3; vals[p] = v.w; } }
    }
    __syncthreads();
    int nnz = cnt < 1024 ? cnt : 1024;
    if (tid == 0) {                       // canonical order -> deterministic fp sum
      for (int a = 1; a < nnz; ++a) {
        int ia = idxs[a]; float va = vals[a]; int bpos = a - 1;
        while (bpos >= 0 && idxs[bpos] > ia) {
          idxs[bpos + 1] = idxs[bpos]; vals[bpos + 1] = vals[bpos]; --bpos;
        }
        idxs[bpos + 1] = ia; vals[bpos + 1] = va;
      }
    }
    __syncthreads();
    // data-dependent gather addresses: prefetch the weight rows (global_prefetch_b8)
    for (int kk = tid; kk < nnz; kk += 320)
      __builtin_prefetch(Wt + (size_t)idxs[kk] * NBASE, 0, 3);
    if (tid < NBASE) {
      float a = bia[tid];
      for (int kk = 0; kk < nnz; ++kk)
        a += vals[kk] * Wt[(size_t)idxs[kk] * NBASE + tid];   // coalesced 1152B rows
      if (pass == 0) accW = a; else accB = a;
    }
    __syncthreads();
  }

  if (tid < NBASE) {
    float p = pov[r];
    float c0 = p * accW + (1.0f - p) * accB;   // cat position tid
    float c1 = p * accB + (1.0f - p) * accW;   // cat position 288+tid
    c0 = c0 > 0.0f ? c0 : 0.0f;
    c1 = c1 > 0.0f ? c1 : 0.0f;
    size_t bo = (size_t)r * NCAT;
    baseF[bo + tid]         = c0;
    baseF[bo + NBASE + tid] = c1;
    size_t xo = (size_t)r * KPAD;
    Xbf[xo + tid]           = (__bf16)c0;
    Xbf[xo + NBASE + tid]   = (__bf16)c1;
  }
}

// ---------------------------------------------------------------------------
// Small bf16 WMMA GEMM (a0 head): direct-from-global fragments.
// C[M,160] = relu(A[M,576] * Wa0b[160,576]^T + b_a0) -> bf16 into X cols 576+.
// Wave tile 64(M)x32(N), 4 waves/block => 256x32 block tile.
// ---------------------------------------------------------------------------
__global__ __launch_bounds__(128)
void wmma_a0_kernel(const __bf16* __restrict__ A, int lda,
                    const __bf16* __restrict__ Bw, int ldb,
                    int K,
                    const float* __restrict__ bias0, int biasN,
                    __bf16* __restrict__ outB, int ldob, int outColOff) {
  const int lane = threadIdx.x & 31;
  const int wv   = threadIdx.x >> 5;
  const int l16  = lane & 15;
  const int hsel = lane >> 4;                  // 0: lanes 0-15, 1: lanes 16-31
  const int mBase = blockIdx.y * 256 + wv * 64;
  const int nBase = blockIdx.x * 32;

  v8f acc[4][2];
  #pragma unroll
  for (int mi = 0; mi < 4; ++mi)
    #pragma unroll
    for (int ni = 0; ni < 2; ++ni)
      #pragma unroll
      for (int i = 0; i < 8; ++i) acc[mi][ni][i] = 0.0f;

  for (int k = 0; k < K; k += 32) {
    v16bf aF[4], bF[2];
    #pragma unroll
    for (int mi = 0; mi < 4; ++mi) {
      const __bf16* pa = A + (size_t)(mBase + mi * 16 + l16) * lda + k + hsel * 16;
      v8bf lo = *(const v8bf*)pa;
      v8bf hi = *(const v8bf*)(pa + 8);
      #pragma unroll
      for (int t = 0; t < 8; ++t) { aF[mi][t] = lo[t]; aF[mi][t + 8] = hi[t]; }
    }
    #pragma unroll
    for (int ni = 0; ni < 2; ++ni) {
      const __bf16* pb = Bw + (size_t)(nBase + ni * 16 + l16) * ldb + k + hsel * 16;
      bF[ni] = *(const v16bf*)pb;
    }
    #pragma unroll
    for (int mi = 0; mi < 4; ++mi)
      #pragma unroll
      for (int ni = 0; ni < 2; ++ni)
        acc[mi][ni] = __builtin_amdgcn_wmma_f32_16x16x32_bf16(
            false, aF[mi], false, bF[ni], (short)0, acc[mi][ni], false, false);
  }

  #pragma unroll
  for (int ni = 0; ni < 2; ++ni) {
    int n = nBase + ni * 16 + l16;
    float b = (n < biasN) ? bias0[n] : 0.0f;
    #pragma unroll
    for (int mi = 0; mi < 4; ++mi) {
      #pragma unroll
      for (int i = 0; i < 8; ++i) {
        int m = mBase + mi * 16 + hsel * 8 + i;
        float c = acc[mi][ni][i] + b;
        c = c > 0.0f ? c : 0.0f;
        outB[(size_t)m * ldob + outColOff + n] = (__bf16)c;
      }
    }
  }
}

// ---------------------------------------------------------------------------
// Big fused action GEMM with CDNA5 async-to-LDS staging of the B tile.
// C[2048,4096] = X[2048,736] * Wcat[4096,736]^T + b_a1 + b_as.
// Block tile 128(M) x 128(N); 4 waves stacked in M, wave tile 32(M)x128(N).
// B tile (128 n-rows x 32 k, bf16) double-buffered in LDS with an 80B row
// pitch (conflict-free ds_load_b128: banks 20n mod 64), filled with
// GLOBAL_LOAD_ASYNC_TO_LDS_B128 and synchronized via s_wait_asynccnt.
// ---------------------------------------------------------------------------
static constexpr int LDS_ROW   = 40;                // bf16 elems per row (80 B)
static constexpr int STAGE_E   = 128 * LDS_ROW;     // elems per stage (10240 B)

__device__ __forceinline__
void stage_b_async(const __bf16* __restrict__ Bw, int nBase, int k0,
                   unsigned ldsBase, int sel, int tid) {
  #pragma unroll
  for (int q = 0; q < 4; ++q) {
    int idx = q * 128 + tid;            // 512 chunks: 128 rows x 4 x 16B
    int row = idx >> 2;
    int ch  = idx & 3;
    const __bf16* g = Bw + (size_t)(nBase + row) * KPAD + k0 + ch * 8;
    unsigned l = ldsBase + (unsigned)(sel * (STAGE_E * 2) + row * (LDS_ROW * 2) + ch * 16);
    asm volatile("global_load_async_to_lds_b128 %0, %1, off"
                 :: "v"(l), "v"(g) : "memory");
  }
}

__global__ __launch_bounds__(128)
void wmma_act_kernel(const __bf16* __restrict__ A,      // X  [2048,736]
                     const __bf16* __restrict__ Bw,     // Wcat [4096,736]
                     const float* __restrict__ bias0,   // b_a1
                     const float* __restrict__ bias1,   // b_as
                     float* __restrict__ outF) {        // act out, ld NMOVE
  __shared__ __align__(64) __bf16 bt[2 * STAGE_E];

  const int tid  = threadIdx.x;
  const int lane = tid & 31;
  const int wv   = tid >> 5;
  const int l16  = lane & 15;
  const int hsel = lane >> 4;
  const int mBase = blockIdx.y * 128 + wv * 32;
  const int nBase = blockIdx.x * 128;

  unsigned ldsBase;
  {
    __attribute__((address_space(3))) __bf16* p3 =
        (__attribute__((address_space(3))) __bf16*)bt;
    ldsBase = (unsigned)(unsigned long long)p3;
  }

  v8f acc[2][8];
  #pragma unroll
  for (int mi = 0; mi < 2; ++mi)
    #pragma unroll
    for (int ni = 0; ni < 8; ++ni)
      #pragma unroll
      for (int i = 0; i < 8; ++i) acc[mi][ni][i] = 0.0f;

  // preload first B stage
  stage_b_async(Bw, nBase, 0, ldsBase, 0, tid);

  for (int k = 0; k < KPAD; k += 32) {
    const int cur = (k >> 5) & 1;
    // my async writes done -> barrier makes every wave's writes visible
    asm volatile("s_wait_asynccnt 0x0" ::: "memory");
    __syncthreads();
    if (k + 32 < KPAD)
      stage_b_async(Bw, nBase, k + 32, ldsBase, cur ^ 1, tid);

    // A fragments direct from global (rows disjoint per wave, coalesced 16B)
    v16bf aF[2];
    #pragma unroll
    for (int mi = 0; mi < 2; ++mi) {
      const __bf16* pa = A + (size_t)(mBase + mi * 16 + l16) * KPAD + k + hsel * 16;
      v8bf lo = *(const v8bf*)pa;
      v8bf hi = *(const v8bf*)(pa + 8);
      #pragma unroll
      for (int t = 0; t < 8; ++t) { aF[mi][t] = lo[t]; aF[mi][t + 8] = hi[t]; }
    }

    #pragma unroll
    for (int ni = 0; ni < 8; ++ni) {
      const __bf16* pb = bt + cur * STAGE_E + (ni * 16 + l16) * LDS_ROW + hsel * 16;
      v16bf bF = *(const v16bf*)pb;                 // 2x ds_load_b128
      acc[0][ni] = __builtin_amdgcn_wmma_f32_16x16x32_bf16(
          false, aF[0], false, bF, (short)0, acc[0][ni], false, false);
      acc[1][ni] = __builtin_amdgcn_wmma_f32_16x16x32_bf16(
          false, aF[1], false, bF, (short)0, acc[1][ni], false, false);
    }
  }

  #pragma unroll
  for (int ni = 0; ni < 8; ++ni) {
    int n = nBase + ni * 16 + l16;                  // C/D: N = lane%16
    float b = bias0[n] + bias1[n];
    #pragma unroll
    for (int mi = 0; mi < 2; ++mi) {
      #pragma unroll
      for (int i = 0; i < 8; ++i) {
        int m = mBase + mi * 16 + hsel * 8 + i;     // C/D: M = vgpr + 8*(lane>=16)
        outF[(size_t)m * NMOVE + n] = acc[mi][ni][i] + b;
      }
    }
  }
}

// ---------------------------------------------------------------------------
// Value head: 576 -> 32 -> 32 -> 1 plus 576 -> 1 skip. One block per row.
// ---------------------------------------------------------------------------
__global__ __launch_bounds__(256)
void vhead_kernel(const float* __restrict__ baseF,
                  const float* __restrict__ W_v0, const float* __restrict__ b_v0,
                  const float* __restrict__ W_v1, const float* __restrict__ b_v1,
                  const float* __restrict__ W_v2, const float* __restrict__ b_v2,
                  const float* __restrict__ W_vs, const float* __restrict__ b_vs,
                  float* __restrict__ outVal) {
  const int r = blockIdx.x, tid = threadIdx.x;
  __shared__ float bs[NCAT];
  __shared__ float v0[32];
  __shared__ float v1[32];
  __shared__ float red[256];

  for (int c = tid; c < NCAT; c += 256) bs[c] = baseF[(size_t)r * NCAT + c];
  __syncthreads();
  if (tid < 32) {
    float s = b_v0[tid];
    for (int k = 0; k < NCAT; ++k) s += W_v0[(size_t)tid * NCAT + k] * bs[k];
    v0[tid] = s > 0.0f ? s : 0.0f;
  }
  __syncthreads();
  if (tid < 32) {
    float s = b_v1[tid];
    for (int k = 0; k < 32; ++k) s += W_v1[tid * 32 + k] * v0[k];
    v1[tid] = s > 0.0f ? s : 0.0f;
  }
  __syncthreads();
  float part = 0.0f;
  for (int c = tid; c < NCAT; c += 256) part += W_vs[c] * bs[c];
  if (tid < 32) part += W_v2[tid] * v1[tid];
  red[tid] = part;
  __syncthreads();
  for (int s = 128; s > 0; s >>= 1) {
    if (tid < s) red[tid] += red[tid + s];
    __syncthreads();
  }
  if (tid == 0) outVal[r] = red[0] + b_v2[0] + b_vs[0];
}

// ---------------------------------------------------------------------------
extern "C" void kernel_launch(void* const* d_in, const int* in_sizes, int n_in,
                              void* d_out, int out_size, void* d_ws, size_t ws_size,
                              hipStream_t stream) {
  const float* pov  = (const float*)d_in[0];
  const float* white= (const float*)d_in[1];
  const float* black= (const float*)d_in[2];
  const float* W_w  = (const float*)d_in[3];
  const float* b_w  = (const float*)d_in[4];
  const float* W_b  = (const float*)d_in[5];
  const float* b_b  = (const float*)d_in[6];
  const float* W_v0 = (const float*)d_in[7];
  const float* b_v0 = (const float*)d_in[8];
  const float* W_v1 = (const float*)d_in[9];
  const float* b_v1 = (const float*)d_in[10];
  const float* W_v2 = (const float*)d_in[11];
  const float* b_v2 = (const float*)d_in[12];
  const float* W_a0 = (const float*)d_in[13];
  const float* b_a0 = (const float*)d_in[14];
  const float* W_a1 = (const float*)d_in[15];
  const float* b_a1 = (const float*)d_in[16];
  const float* W_vs = (const float*)d_in[17];
  const float* b_vs = (const float*)d_in[18];
  const float* W_as = (const float*)d_in[19];
  const float* b_as = (const float*)d_in[20];

  // workspace layout (bytes)
  char* ws = (char*)d_ws;
  float*  WtW   = (float*) (ws + 0);          // 40960*288*4 = 47,185,920
  float*  WtB   = (float*) (ws + 47185920);   // 47,185,920
  float*  baseF = (float*) (ws + 94371840);   // 2048*576*4 = 4,718,592
  __bf16* Xbf   = (__bf16*)(ws + 99090432);   // 2048*736*2 = 3,014,656
  __bf16* Wcat  = (__bf16*)(ws + 102105088);  // 4096*736*2 = 6,029,312
  __bf16* Wa0b  = (__bf16*)(ws + 108134400);  // 160*576*2  =   184,320

  float* outVal = (float*)d_out;              // [2048]
  float* outAct = (float*)d_out + BATCH;      // [2048*4096]

  // 1. transpose accumulator weights (independent)
  transpose_kernel<<<dim3(D_IN / 32, NBASE / 32), dim3(32, 8), 0, stream>>>(W_w, WtW);
  transpose_kernel<<<dim3(D_IN / 32, NBASE / 32), dim3(32, 8), 0, stream>>>(W_b, WtB);
  // 2. pack bf16 weights for WMMA (independent)
  pack_cat_kernel<<<((size_t)NMOVE * KPAD + 255) / 256, 256, 0, stream>>>(W_as, W_a1, Wcat);
  pack_wa0_kernel<<<((size_t)NA0P * NCAT + 255) / 256, 256, 0, stream>>>(W_a0, Wa0b);
  // 3. sparse accumulator -> base (f32) + X[:,0:576] (bf16)
  accum_kernel<<<BATCH, 320, 0, stream>>>(pov, white, black, WtW, b_w, WtB, b_b, baseF, Xbf);
  // 4. a0 head via WMMA: relu(base @ W_a0^T + b_a0) -> X[:,576:736] (padded rows give 0s)
  wmma_a0_kernel<<<dim3(NA0P / 32, BATCH / 256), 128, 0, stream>>>(
      Xbf, KPAD, Wa0b, NCAT, NCAT, b_a0, 144, Xbf, KPAD, NCAT);
  // 5. fused action GEMM via WMMA + async-to-LDS B staging
  wmma_act_kernel<<<dim3(NMOVE / 128, BATCH / 128), 128, 0, stream>>>(
      Xbf, Wcat, b_a1, b_as, outAct);
  // 6. value head
  vhead_kernel<<<BATCH, 256, 0, stream>>>(baseF, W_v0, b_v0, W_v1, b_v1,
                                          W_v2, b_v2, W_vs, b_vs, outVal);
}
